// ForecastNet_39659728011614
// MI455X (gfx1250) — matrix operand order repaired and checked by reading, not compiled, not run on hardware
//
#include <hip/hip_runtime.h>
#include <cstdint>
#include <cstddef>

#define B_   256
#define T_   168
#define F_   8
#define H_   512
#define G4H_ (4 * H_)
#define OUT_ 24

typedef __attribute__((ext_vector_type(16))) __bf16 v16bf;
typedef __attribute__((ext_vector_type(8)))  float  v8f;

struct U32x8 { uint4 lo, hi; };

static __device__ __forceinline__ v16bf ldbf16x16(const __bf16* p0, const __bf16* p1) {
  U32x8 u;
  u.lo = *reinterpret_cast<const uint4*>(p0);
  u.hi = *reinterpret_cast<const uint4*>(p1);
  return __builtin_bit_cast(v16bf, u);
}

static __device__ __forceinline__ float sigmoidf_(float x) {
  return 1.0f / (1.0f + __expf(-x));
}
static __device__ __forceinline__ float tanhf_(float x) {
  float e = __expf(-2.0f * x);
  return (1.0f - e) / (1.0f + e);
}

// Accumulate 2 row-tiles x 4 gate-tiles (each 16x16) over K=512.
// acc[r][g]: r = row-half (rows m0+r*16 .. +16), g = gate.
// Each B tile (32Kx16N slice of the weight matrix) is loaded once and feeds
// two WMMAs (both A row tiles) -> 12 b128 loads per 8 WMMAs, and 8 independent
// accumulation chains per wave for XDL latency hiding.
static __device__ __forceinline__ void gemm8_k512(
    const __bf16* __restrict__ Arow0,   // lane's A row in rows [m0, m0+16)
    const __bf16* __restrict__ Arow1,   // lane's A row in rows [m0+16, m0+32)
    const __bf16* __restrict__ Bmat,    // (4*H_) x H_ bf16 row-major
    int ncol, int akoff, int bkoff, v8f acc[2][4])
{
#pragma unroll 2
  for (int k0 = 0; k0 < H_; k0 += 32) {
    v16bf a0 = ldbf16x16(Arow0 + k0 + akoff, Arow0 + k0 + akoff + 16);
    v16bf a1 = ldbf16x16(Arow1 + k0 + akoff, Arow1 + k0 + akoff + 16);
#pragma unroll
    for (int g = 0; g < 4; ++g) {
      const __bf16* bp = Bmat + (size_t)(g * H_ + ncol) * H_ + k0 + bkoff;
      v16bf bv = ldbf16x16(bp, bp + 16);
      acc[0][g] = __builtin_amdgcn_wmma_f32_16x16x32_bf16(
          false, a0, false, bv, (short)0, acc[0][g], false, false);
      acc[1][g] = __builtin_amdgcn_wmma_f32_16x16x32_bf16(
          false, a1, false, bv, (short)0, acc[1][g], false, false);
    }
  }
}

// One LSTM cell step for the whole batch.
// Workgroup: 64 threads = 2 waves; wave w owns cols [blockIdx.x*32 + w*16, +16)
// for all 4 gates; rows [blockIdx.y*32, +32). Grid: (H_/32, B_/32) = (16, 8)
// -> 128 workgroups spread across WGPs.
// BIG_IN: input is B_ x H_ bf16 (second WMMA GEMM). !BIG_IN: input is B_ x 8
// f32 with row stride xstride (VALU dot products).
template <bool BIG_IN>
__global__ __launch_bounds__(64) void lstm_step_kernel(
    const __bf16* __restrict__ hprev,    // B_ x H_ bf16
    const __bf16* __restrict__ xbf,      // B_ x H_ bf16 (BIG_IN)
    const float*  __restrict__ xf32,     // B_ x F_ f32 (!BIG_IN), stride xstride
    int xstride,
    const __bf16* __restrict__ Whh,      // 4H x H bf16 row-major
    const __bf16* __restrict__ Wih_bf,   // 4H x H bf16 row-major (BIG_IN)
    const float*  __restrict__ Wih_f32,  // 4H x 8 f32 (!BIG_IN)
    const float*  __restrict__ bias,     // 4H f32 (b_ih + b_hh)
    float* __restrict__ c,               // B_ x H_ f32 (in/out)
    __bf16* __restrict__ hout)           // B_ x H_ bf16
{
  const int lane  = threadIdx.x & 31;
  const int wave  = threadIdx.x >> 5;
  const int m0    = blockIdx.y * 32;
  const int n0    = blockIdx.x * 32 + wave * 16;
  const int lhalf = lane >> 4;
  const int l15   = lane & 15;
  const int arow  = m0 + l15;       // A-matrix row held by this lane (tile 0)
  const int akoff = lhalf * 8;      // A-matrix K sub-offset (ISA 16-bit A layout)
  const int ncol  = n0 + l15;       // output column / B-matrix column
  const int bkoff = lhalf * 8;      // B-matrix K sub-offset (same layout as A)

  // C starts as the bias (depends only on column -> same for all D rows)
  v8f acc[2][4];
  {
    float b0 = bias[0 * H_ + ncol];
    float b1 = bias[1 * H_ + ncol];
    float b2 = bias[2 * H_ + ncol];
    float b3 = bias[3 * H_ + ncol];
#pragma unroll
    for (int r = 0; r < 2; ++r)
#pragma unroll
      for (int vi = 0; vi < 8; ++vi) {
        acc[r][0][vi] = b0; acc[r][1][vi] = b1;
        acc[r][2][vi] = b2; acc[r][3][vi] = b3;
      }
  }

  // Recurrent GEMM: h_prev @ Whh^T
  gemm8_k512(hprev + (size_t)arow * H_, hprev + (size_t)(arow + 16) * H_,
             Whh, ncol, akoff, bkoff, acc);

  if (BIG_IN) {
    // Input GEMM: x @ Wih^T (K = 512)
    gemm8_k512(xbf + (size_t)arow * H_, xbf + (size_t)(arow + 16) * H_,
               Wih_bf, ncol, akoff, bkoff, acc);
  } else {
    // Tiny input (Din = 8): plain FMA dot products in f32
    float wih[4][8];
#pragma unroll
    for (int g = 0; g < 4; ++g)
#pragma unroll
      for (int k = 0; k < 8; ++k)
        wih[g][k] = Wih_f32[(size_t)(g * H_ + ncol) * 8 + k];
#pragma unroll
    for (int r = 0; r < 2; ++r)
#pragma unroll
      for (int vi = 0; vi < 8; ++vi) {
        const int row = m0 + r * 16 + vi + lhalf * 8;
        const float* xp = xf32 + (size_t)row * xstride;
        float xv[8];
#pragma unroll
        for (int k = 0; k < 8; ++k) xv[k] = xp[k];
#pragma unroll
        for (int g = 0; g < 4; ++g) {
          float s = 0.0f;
#pragma unroll
          for (int k = 0; k < 8; ++k) s = __fmaf_rn(xv[k], wih[g][k], s);
          acc[r][g][vi] += s;
        }
      }
  }

  // Fused gates + cell/hidden update, all in-register (i,f,g,o for a given
  // (b,n) live in the same lane / same VGPR slot of the four accumulators).
#pragma unroll
  for (int r = 0; r < 2; ++r)
#pragma unroll
    for (int vi = 0; vi < 8; ++vi) {
      const int row = m0 + r * 16 + vi + lhalf * 8;  // D layout: M = vi + 8*(lane>>4)
      const int idx = row * H_ + ncol;
      float gi = acc[r][0][vi];
      float gf = acc[r][1][vi];
      float gg = acc[r][2][vi];
      float go = acc[r][3][vi];
      float is = sigmoidf_(gi);
      float fs = sigmoidf_(gf);
      float gt = tanhf_(gg);
      float os = sigmoidf_(go);
      float cn = __fmaf_rn(fs, c[idx], is * gt);
      c[idx] = cn;
      float hv = os * tanhf_(cn);
      hout[idx] = (__bf16)hv;
    }
}

__global__ __launch_bounds__(256) void f32_to_bf16_kernel(
    const float* __restrict__ src, __bf16* __restrict__ dst, int n) {
  int i = blockIdx.x * 256 + threadIdx.x;
  if (i < n) dst[i] = (__bf16)src[i];
}

__global__ __launch_bounds__(256) void bias_add_kernel(
    const float* __restrict__ a, const float* __restrict__ b,
    float* __restrict__ o, int n) {
  int i = blockIdx.x * 256 + threadIdx.x;
  if (i < n) o[i] = a[i] + b[i];
}

// pred = h1 @ fcW^T + fcb; also scatter into d_out[b][step][f]
__global__ __launch_bounds__(256) void fc_kernel(
    const __bf16* __restrict__ h1, const float* __restrict__ W,
    const float* __restrict__ bvec, float* __restrict__ pred,
    float* __restrict__ out, int step) {
  int t = blockIdx.x * 256 + threadIdx.x;
  if (t >= B_ * F_) return;
  int b = t >> 3, f = t & 7;
  const __bf16* hp = h1 + (size_t)b * H_;
  const float*  wp = W + (size_t)f * H_;
  float s = bvec[f];
  for (int k = 0; k < H_; ++k) s = __fmaf_rn((float)hp[k], wp[k], s);
  pred[t] = s;
  out[(size_t)b * (OUT_ * F_) + step * F_ + f] = s;
}

extern "C" void kernel_launch(void* const* d_in, const int* in_sizes, int n_in,
                              void* d_out, int out_size, void* d_ws, size_t ws_size,
                              hipStream_t stream) {
  (void)in_sizes; (void)n_in; (void)out_size; (void)ws_size;
  const float* src    = (const float*)d_in[0];
  const float* eWih0  = (const float*)d_in[1];
  const float* eWhh0  = (const float*)d_in[2];
  const float* eb_ih0 = (const float*)d_in[3];
  const float* eb_hh0 = (const float*)d_in[4];
  const float* eWih1  = (const float*)d_in[5];
  const float* eWhh1  = (const float*)d_in[6];
  const float* eb_ih1 = (const float*)d_in[7];
  const float* eb_hh1 = (const float*)d_in[8];
  const float* dWih0  = (const float*)d_in[9];
  const float* dWhh0  = (const float*)d_in[10];
  const float* db_ih0 = (const float*)d_in[11];
  const float* db_hh0 = (const float*)d_in[12];
  const float* dWih1  = (const float*)d_in[13];
  const float* dWhh1  = (const float*)d_in[14];
  const float* db_ih1 = (const float*)d_in[15];
  const float* db_hh1 = (const float*)d_in[16];
  const float* fcW    = (const float*)d_in[17];
  const float* fcb    = (const float*)d_in[18];
  float* out = (float*)d_out;

  // ---- workspace bump allocator (≈56 MB total) ----
  char* ws = (char*)d_ws;
  size_t off = 0;
  auto alloc = [&](size_t bytes) -> void* {
    void* p = ws + off;
    off = (off + bytes + 255) & ~(size_t)255;
    return p;
  };
  const size_t WB = (size_t)G4H_ * H_ * sizeof(__bf16);  // 2 MB per weight matrix
  __bf16* eWhh0_bf = (__bf16*)alloc(WB);
  __bf16* eWih1_bf = (__bf16*)alloc(WB);
  __bf16* eWhh1_bf = (__bf16*)alloc(WB);
  __bf16* dWhh0_bf = (__bf16*)alloc(WB);
  __bf16* dWih1_bf = (__bf16*)alloc(WB);
  __bf16* dWhh1_bf = (__bf16*)alloc(WB);
  float* bias_e0 = (float*)alloc(G4H_ * sizeof(float));
  float* bias_e1 = (float*)alloc(G4H_ * sizeof(float));
  float* bias_d0 = (float*)alloc(G4H_ * sizeof(float));
  float* bias_d1 = (float*)alloc(G4H_ * sizeof(float));
  const size_t HB = (size_t)B_ * H_ * sizeof(__bf16);    // 256 KB
  __bf16* y0    = (__bf16*)alloc((size_t)T_ * HB);        // layer-0 outputs, all t (~42 MB)
  __bf16* h1    = (__bf16*)alloc(2 * HB);                 // enc layer-1 ping-pong
  __bf16* hd0   = (__bf16*)alloc(2 * HB);                 // dec layer-0 ping-pong
  __bf16* hd1   = (__bf16*)alloc(2 * HB);                 // dec layer-1 ping-pong
  __bf16* hzero = (__bf16*)alloc(HB);
  float*  c0    = (float*)alloc((size_t)B_ * H_ * sizeof(float));
  float*  c1    = (float*)alloc((size_t)B_ * H_ * sizeof(float));
  float*  pred  = (float*)alloc((size_t)B_ * F_ * sizeof(float));

  // ---- weight conversion (cheap; runs every call for determinism) ----
  const int WN = G4H_ * H_;
  dim3 cg((WN + 255) / 256), cb(256);
  f32_to_bf16_kernel<<<cg, cb, 0, stream>>>(eWhh0, eWhh0_bf, WN);
  f32_to_bf16_kernel<<<cg, cb, 0, stream>>>(eWih1, eWih1_bf, WN);
  f32_to_bf16_kernel<<<cg, cb, 0, stream>>>(eWhh1, eWhh1_bf, WN);
  f32_to_bf16_kernel<<<cg, cb, 0, stream>>>(dWhh0, dWhh0_bf, WN);
  f32_to_bf16_kernel<<<cg, cb, 0, stream>>>(dWih1, dWih1_bf, WN);
  f32_to_bf16_kernel<<<cg, cb, 0, stream>>>(dWhh1, dWhh1_bf, WN);
  dim3 bg((G4H_ + 255) / 256);
  bias_add_kernel<<<bg, cb, 0, stream>>>(eb_ih0, eb_hh0, bias_e0, G4H_);
  bias_add_kernel<<<bg, cb, 0, stream>>>(eb_ih1, eb_hh1, bias_e1, G4H_);
  bias_add_kernel<<<bg, cb, 0, stream>>>(db_ih0, db_hh0, bias_d0, G4H_);
  bias_add_kernel<<<bg, cb, 0, stream>>>(db_ih1, db_hh1, bias_d1, G4H_);
  hipMemsetAsync(hzero, 0, HB, stream);
  hipMemsetAsync(c0, 0, (size_t)B_ * H_ * sizeof(float), stream);
  hipMemsetAsync(c1, 0, (size_t)B_ * H_ * sizeof(float), stream);

  const dim3 sg(H_ / 32, B_ / 32);  // (16, 8) -> 128 WGs x 2 waves
  const dim3 sb(64);

  // ---- encoder layer 0 (Din=8 f32 input, WMMA recurrent) ----
  for (int t = 0; t < T_; ++t) {
    const __bf16* hp = (t == 0) ? hzero : (y0 + (size_t)(t - 1) * B_ * H_);
    lstm_step_kernel<false><<<sg, sb, 0, stream>>>(
        hp, nullptr, src + (size_t)t * F_, T_ * F_,
        eWhh0_bf, nullptr, eWih0, bias_e0, c0,
        y0 + (size_t)t * B_ * H_);
  }
  // ---- encoder layer 1 (two WMMA GEMMs per step) ----
  for (int t = 0; t < T_; ++t) {
    const __bf16* hp = (t == 0) ? hzero : (h1 + (size_t)((t - 1) & 1) * B_ * H_);
    lstm_step_kernel<true><<<sg, sb, 0, stream>>>(
        hp, y0 + (size_t)t * B_ * H_, nullptr, 0,
        eWhh1_bf, eWih1_bf, nullptr, bias_e1, c1,
        h1 + (size_t)(t & 1) * B_ * H_);
  }

  const __bf16* henc0 = y0 + (size_t)(T_ - 1) * B_ * H_;
  const __bf16* henc1 = h1 + (size_t)((T_ - 1) & 1) * B_ * H_;

  // ---- decoder: 24 autoregressive steps (c0/c1 continue from encoder) ----
  for (int s = 0; s < OUT_; ++s) {
    const __bf16* hp0 = (s == 0) ? henc0 : (hd0 + (size_t)((s - 1) & 1) * B_ * H_);
    const float*  xin = (s == 0) ? (src + (size_t)(T_ - 1) * F_) : pred;
    const int     xs  = (s == 0) ? (T_ * F_) : F_;
    lstm_step_kernel<false><<<sg, sb, 0, stream>>>(
        hp0, nullptr, xin, xs,
        dWhh0_bf, nullptr, dWih0, bias_d0, c0,
        hd0 + (size_t)(s & 1) * B_ * H_);
    const __bf16* hp1 = (s == 0) ? henc1 : (hd1 + (size_t)((s - 1) & 1) * B_ * H_);
    lstm_step_kernel<true><<<sg, sb, 0, stream>>>(
        hp1, hd0 + (size_t)(s & 1) * B_ * H_, nullptr, 0,
        dWhh1_bf, dWih1_bf, nullptr, bias_d1, c1,
        hd1 + (size_t)(s & 1) * B_ * H_);
    fc_kernel<<<(B_ * F_ + 255) / 256, 256, 0, stream>>>(
        hd1 + (size_t)(s & 1) * B_ * H_, fcW, fcb, pred, out, s);
  }
}
